// Model_2010044695209
// MI455X (gfx1250) — matrix operand order repaired and checked by reading, not compiled, not run on hardware
//
#include <hip/hip_runtime.h>
#include <hip/hip_bf16.h>

// ---------------------------------------------------------------------------
// Problem constants (match reference)
// ---------------------------------------------------------------------------
static constexpr int S_LEN = 4096;    // sequence length
static constexpr int DIM   = 1024;    // input dim == 2*H
static constexpr int H     = 512;     // per-direction hidden
static constexpr int G4H   = 2048;    // 4*H gate rows
static constexpr int TTAGS = 50;      // tags
static constexpr int TPAD  = 64;      // padded tag dim for WMMA
static constexpr int SOS_T = TTAGS - 2;
static constexpr int EOS_T = TTAGS - 1;

// ---------------------------------------------------------------------------
// CDNA5 WMMA types
// ---------------------------------------------------------------------------
typedef __attribute__((ext_vector_type(16))) __bf16 v16bf;
typedef __attribute__((ext_vector_type(8)))  float  v8f;
typedef __attribute__((ext_vector_type(4)))  unsigned int u32x4;

// ---------------------------------------------------------------------------
// Helpers
// ---------------------------------------------------------------------------
__device__ __forceinline__ unsigned short f2bf(float f) {
    union { float f; unsigned u; } x; x.f = f;
    unsigned r = x.u + 0x7fffu + ((x.u >> 16) & 1u);   // round-to-nearest-even
    return (unsigned short)(r >> 16);
}
__device__ __forceinline__ float bflo(unsigned v) {
    union { unsigned u; float f; } x; x.u = v << 16; return x.f;
}
__device__ __forceinline__ float bfhi(unsigned v) {
    union { unsigned u; float f; } x; x.u = v & 0xffff0000u; return x.f;
}
__device__ __forceinline__ float dot8(u32x4 u, const float* h) {
    return bflo(u.x) * h[0] + bfhi(u.x) * h[1] +
           bflo(u.y) * h[2] + bfhi(u.y) * h[3] +
           bflo(u.z) * h[4] + bfhi(u.z) * h[5] +
           bflo(u.w) * h[6] + bfhi(u.w) * h[7];
}
__device__ __forceinline__ float sigmoidf(float x) { return 1.0f / (1.0f + expf(-x)); }

// ---------------------------------------------------------------------------
// Small prep kernels
// ---------------------------------------------------------------------------
__global__ void cvt_bf16_kernel(const float* __restrict__ src,
                                unsigned short* __restrict__ dst, int n) {
    for (int i = blockIdx.x * blockDim.x + threadIdx.x; i < n;
         i += gridDim.x * blockDim.x)
        dst[i] = f2bf(src[i]);
}

__global__ void bias_sum_kernel(const float* __restrict__ a,
                                const float* __restrict__ b,
                                float* __restrict__ c, int n) {
    int i = blockIdx.x * blockDim.x + threadIdx.x;
    if (i < n) c[i] = a[i] + b[i];
}

// pad W_aff [50][1024] -> bf16 [64][1024] (zero rows 50..63), b_aff -> [64]
__global__ void pad_affine_kernel(const float* __restrict__ W,
                                  const float* __restrict__ b,
                                  unsigned short* __restrict__ Wp,
                                  float* __restrict__ bp) {
    int i = blockIdx.x * blockDim.x + threadIdx.x;   // 64*1024 threads
    int r = i >> 10, d = i & 1023;
    Wp[i] = (r < TTAGS) ? f2bf(W[r * DIM + d]) : (unsigned short)0;
    if (i < TPAD) bp[i] = (i < TTAGS) ? b[i] : 0.0f;
}

// x0[s][d] = bf16(embed[words[s]][d])
__global__ void gather_embed_kernel(const int* __restrict__ words,
                                    const float* __restrict__ embed,
                                    unsigned short* __restrict__ x0) {
    size_t i = (size_t)blockIdx.x * blockDim.x + threadIdx.x;   // S*DIM threads
    int s = (int)(i >> 10), d = (int)(i & 1023);
    x0[i] = f2bf(embed[(size_t)words[s] * DIM + d]);
}

// ---------------------------------------------------------------------------
// WMMA GEMM:  C[m][n] = sum_k A[m][k] * Bw[n][k] + bias[n]
//   A : bf16 [tilesM*16][K] row-major,  Bw : bf16 [tilesN*16][K] row-major,
//   C : f32  [tilesM*16][N] row-major (N == tilesN*16).
//   blockIdx.y = dir with element strides for Bw / bias / C.
// One wave computes one 16x16 tile via v_wmma_f32_16x16x32_bf16.
// ---------------------------------------------------------------------------
__global__ void wmma_gemm_bias_kernel(const unsigned short* __restrict__ A,
                                      const unsigned short* __restrict__ Bw,
                                      const float* __restrict__ bias,
                                      float* __restrict__ C,
                                      int K, int N, int tilesM, int tilesN,
                                      long long bDirStride,
                                      long long biasDirStride,
                                      long long cDirStride) {
    const int dir = blockIdx.y;
    const unsigned short* Bp    = Bw   + (size_t)dir * (size_t)bDirStride;
    const float*          biasp = bias + (size_t)dir * (size_t)biasDirStride;
    float*                Cp    = C    + (size_t)dir * (size_t)cDirStride;

    const int wave = threadIdx.x >> 5;
    const int lane = threadIdx.x & 31;
    const long long tileId = (long long)blockIdx.x * (blockDim.x >> 5) + wave;
    if (tileId >= (long long)tilesM * tilesN) return;
    const int tm = (int)(tileId / tilesN);
    const int tn = (int)(tileId % tilesN);

    const int half = lane >> 4;          // 0: lanes 0-15, 1: lanes 16-31
    const int mn   = lane & 15;          // A-row within tile / B-col within tile

    // Accumulator init = bias[col] (ISA C/D layout: column = lane&15)
    const float bcol = biasp[tn * 16 + mn];
    v8f acc;
#pragma unroll
    for (int r = 0; r < 8; ++r) acc[r] = bcol;

    // A 16x32 bf16 tile: lane holds K = (half*8 .. +7) and (half*8+16 .. +23)
    const unsigned short* arow = A  + ((size_t)(tm * 16 + mn)) * K + half * 8;
    // B 32x16 bf16 tile: lane holds K = (half*8 .. +7) and (half*8+16 .. +23) of column mn
    const unsigned short* brow = Bp + ((size_t)(tn * 16 + mn)) * K + half * 8;

    for (int kb = 0; kb < K; kb += 32) {
        union { v16bf v; u32x4 q[2]; } av, bv;
        av.q[0] = *(const u32x4*)(arow);        // 8 bf16: K=half*8..+7
        av.q[1] = *(const u32x4*)(arow + 16);   // 8 bf16: K=half*8+16..+23
        bv.q[0] = *(const u32x4*)(brow);        // 8 bf16: K=half*8..+7
        bv.q[1] = *(const u32x4*)(brow + 16);   // 8 bf16: K=half*8+16..+23
        acc = __builtin_amdgcn_wmma_f32_16x16x32_bf16(
                  false, av.v, false, bv.v, (short)0, acc, false, false);
        arow += 32; brow += 32;
    }

    // D layout: VGPR r -> M = r + 8*half, N = mn
    float* crow = Cp + ((size_t)(tm * 16 + half * 8)) * N + tn * 16 + mn;
#pragma unroll
    for (int r = 0; r < 8; ++r) crow[(size_t)r * N] = acc[r];
}

// ---------------------------------------------------------------------------
// Recurrent scan: one block (1024 threads = 32 waves, one WGP) per direction.
//   gpre : f32 [2][S][2048]  (Wi@x + b_ih + b_hh, gate order i,f,g,o)
//   whh  : bf16 [2][2048][512]   (this layer)
//   h0,c0: f32 [2][512]          (this layer: index 2l+dir)
//   xout : bf16 [S][1024]        (dir 0 -> cols 0..511, dir 1 -> cols 512..1023)
// Thread t<512 owns gate rows (i=j, g=1024+j) and state c[j],h[j];
// thread t>=512 owns (f=512+j, o=1536+j). h lives in LDS; 2 barriers/step.
// ---------------------------------------------------------------------------
__global__ void lstm_scan_kernel(const float* __restrict__ gpre,
                                 const unsigned short* __restrict__ whh,
                                 const float* __restrict__ h0,
                                 const float* __restrict__ c0,
                                 unsigned short* __restrict__ xout) {
    const int dir = blockIdx.x;
    const int t   = threadIdx.x;
    const int j   = t & (H - 1);
    const bool lower = (t < H);

    __shared__ float h_sh[H];
    __shared__ float f_sh[H];
    __shared__ float o_sh[H];

    const int r0 = lower ? j            : H + j;        // i-row / f-row
    const int r1 = lower ? 2 * H + j    : 3 * H + j;    // g-row / o-row
    const unsigned short* w0 = whh + ((size_t)dir * G4H + r0) * H;
    const unsigned short* w1 = whh + ((size_t)dir * G4H + r1) * H;
    const float* gp = gpre + (size_t)dir * S_LEN * G4H;

    float c_reg = lower ? c0[dir * H + j] : 0.0f;
    if (t < H) h_sh[t] = h0[dir * H + t];
    __syncthreads();

    for (int step = 0; step < S_LEN; ++step) {
        const int s = dir ? (S_LEN - 1 - step) : step;
        const float* grow = gp + (size_t)s * G4H;
        float a0 = grow[r0];
        float a1 = grow[r1];
#pragma unroll 4
        for (int k = 0; k < H; k += 8) {
            u32x4 u0 = *(const u32x4*)(w0 + k);
            u32x4 u1 = *(const u32x4*)(w1 + k);
            a0 += dot8(u0, &h_sh[k]);
            a1 += dot8(u1, &h_sh[k]);
        }
        if (!lower) { f_sh[j] = a0; o_sh[j] = a1; }
        __syncthreads();                 // dots done (h_sh free) + f/o visible
        if (lower) {
            const float iv = sigmoidf(a0);
            const float gv = tanhf(a1);
            const float fv = sigmoidf(f_sh[j]);
            const float ov = sigmoidf(o_sh[j]);
            c_reg = fv * c_reg + iv * gv;
            const float hv = ov * tanhf(c_reg);
            h_sh[j] = hv;
            xout[(size_t)s * DIM + dir * H + j] = f2bf(hv);
        }
        __syncthreads();                 // new h visible for next step
    }
}

// ---------------------------------------------------------------------------
// Viterbi decode (single workgroup, T=50).
//   emis : f32 [S][64] (cols >= 50 unused), trans : f32 [50][50]
//   bp   : int [S][64] backpointers, out : f32 [1 + S] = {score, path...}
// ---------------------------------------------------------------------------
__global__ void viterbi_kernel(const float* __restrict__ emis,
                               const float* __restrict__ trans,
                               int* __restrict__ bp,
                               float* __restrict__ out) {
    __shared__ float fv[TPAD];
    __shared__ float nfv[TPAD];
    const int j = threadIdx.x;           // 64 threads

    fv[j] = (j >= TTAGS) ? -3.0e38f : ((j == SOS_T) ? 0.0f : -10000.0f);
    __syncthreads();

    for (int s = 0; s < S_LEN; ++s) {
        if (j < TTAGS) {
            float best = -3.4e38f; int barg = 0;
            for (int p = 0; p < TTAGS; ++p) {
                const float v = fv[p] + trans[j * TTAGS + p];
                if (v > best) { best = v; barg = p; }
            }
            nfv[j] = best + emis[(size_t)s * TPAD + j];
            bp[(size_t)s * TPAD + j] = barg;
        }
        __syncthreads();
        if (j < TTAGS) fv[j] = nfv[j];
        __syncthreads();
    }

    if (j == 0) {
        float best = -3.4e38f; int bt = 0;
        for (int p = 0; p < TTAGS; ++p) {
            const float v = fv[p] + trans[EOS_T * TTAGS + p];
            if (v > best) { best = v; bt = p; }
        }
        out[0] = best;
        int tag = bt;
        for (int t = S_LEN - 1; t >= 0; --t) {
            out[1 + t] = (float)tag;
            tag = bp[(size_t)t * TPAD + tag];
        }
    }
}

// ---------------------------------------------------------------------------
// Workspace layout (bytes, all 256-aligned)
// ---------------------------------------------------------------------------
static constexpr size_t OFF_WIH  = 0;                       // bf16 2*2*2048*1024
static constexpr size_t OFF_WHH  = OFF_WIH  + 16777216;     // bf16 2*2*2048*512
static constexpr size_t OFF_WAFF = OFF_WHH  + 8388608;      // bf16 64*1024
static constexpr size_t OFF_BAFF = OFF_WAFF + 131072;       // f32 64
static constexpr size_t OFF_BSUM = OFF_BAFF + 256;          // f32 2*2*2048
static constexpr size_t OFF_X0   = OFF_BSUM + 32768;        // bf16 S*1024
static constexpr size_t OFF_XA   = OFF_X0   + 8388608;      // bf16 S*1024
static constexpr size_t OFF_XB   = OFF_XA   + 8388608;      // bf16 S*1024
static constexpr size_t OFF_GPRE = OFF_XB   + 8388608;      // f32 2*S*2048
static constexpr size_t OFF_EMIS = OFF_GPRE + 67108864;     // f32 S*64
static constexpr size_t OFF_BP   = OFF_EMIS + 1048576;      // int S*64

extern "C" void kernel_launch(void* const* d_in, const int* in_sizes, int n_in,
                              void* d_out, int out_size, void* d_ws, size_t ws_size,
                              hipStream_t stream) {
    (void)in_sizes; (void)n_in; (void)out_size; (void)ws_size;

    const int*   words = (const int*)  d_in[0];
    const float* embed = (const float*)d_in[1];
    const float* W_ih  = (const float*)d_in[2];   // [2][2][2048][1024]
    const float* W_hh  = (const float*)d_in[3];   // [2][2][2048][512]
    const float* b_ih  = (const float*)d_in[4];   // [2][2][2048]
    const float* b_hh  = (const float*)d_in[5];
    const float* h0    = (const float*)d_in[6];   // [4][512]
    const float* c0    = (const float*)d_in[7];
    const float* W_aff = (const float*)d_in[8];   // [50][1024]
    const float* b_aff = (const float*)d_in[9];   // [50]
    const float* trans = (const float*)d_in[10];  // [50][50]

    char* ws = (char*)d_ws;
    unsigned short* wih_bf  = (unsigned short*)(ws + OFF_WIH);
    unsigned short* whh_bf  = (unsigned short*)(ws + OFF_WHH);
    unsigned short* waff_bf = (unsigned short*)(ws + OFF_WAFF);
    float*          baff    = (float*)(ws + OFF_BAFF);
    float*          bsum    = (float*)(ws + OFF_BSUM);
    unsigned short* x0      = (unsigned short*)(ws + OFF_X0);
    unsigned short* xA      = (unsigned short*)(ws + OFF_XA);
    unsigned short* xB      = (unsigned short*)(ws + OFF_XB);
    float*          gpre    = (float*)(ws + OFF_GPRE);
    float*          emis    = (float*)(ws + OFF_EMIS);
    int*            bpb     = (int*)(ws + OFF_BP);

    // --- weight prep ---
    cvt_bf16_kernel<<<2048, 256, 0, stream>>>(W_ih, wih_bf, 2 * 2 * G4H * DIM);
    cvt_bf16_kernel<<<2048, 256, 0, stream>>>(W_hh, whh_bf, 2 * 2 * G4H * H);
    pad_affine_kernel<<<(TPAD * DIM) / 256, 256, 0, stream>>>(W_aff, b_aff, waff_bf, baff);
    bias_sum_kernel<<<(2 * 2 * G4H) / 256, 256, 0, stream>>>(b_ih, b_hh, bsum, 2 * 2 * G4H);
    gather_embed_kernel<<<(S_LEN * DIM) / 256, 256, 0, stream>>>(words, embed, x0);

    const int tilesM = S_LEN / 16;          // 256
    const int tilesN = G4H / 16;            // 128
    const int gemmBlocks = tilesM * tilesN / 8;   // 8 waves (tiles) per 256-thr block

    // --- layer 0 ---
    wmma_gemm_bias_kernel<<<dim3(gemmBlocks, 2), 256, 0, stream>>>(
        x0, wih_bf, bsum, gpre, DIM, G4H, tilesM, tilesN,
        (long long)G4H * DIM, (long long)G4H, (long long)S_LEN * G4H);
    lstm_scan_kernel<<<2, 1024, 0, stream>>>(gpre, whh_bf, h0, c0, xA);

    // --- layer 1 ---
    wmma_gemm_bias_kernel<<<dim3(gemmBlocks, 2), 256, 0, stream>>>(
        xA, wih_bf + (size_t)2 * G4H * DIM, bsum + 2 * G4H, gpre,
        DIM, G4H, tilesM, tilesN,
        (long long)G4H * DIM, (long long)G4H, (long long)S_LEN * G4H);
    lstm_scan_kernel<<<2, 1024, 0, stream>>>(
        gpre, whh_bf + (size_t)2 * G4H * H, h0 + 2 * H, c0 + 2 * H, xB);

    // --- affine: emission [S][64] ---
    wmma_gemm_bias_kernel<<<dim3(tilesM * (TPAD / 16) / 8, 1), 256, 0, stream>>>(
        xB, waff_bf, baff, emis, DIM, TPAD, tilesM, TPAD / 16, 0LL, 0LL, 0LL);

    // --- Viterbi decode ---
    viterbi_kernel<<<1, TPAD, 0, stream>>>(emis, trans, bpb, (float*)d_out);
}
